// MatchingNetwork_14499809591651
// MI455X (gfx1250) — matrix operand order repaired and checked by reading, not compiled, not run on hardware
//
#include <hip/hip_runtime.h>
#include <hip/hip_bf16.h>

typedef __attribute__((ext_vector_type(16))) _Float16 v16h;
typedef __attribute__((ext_vector_type(8)))  _Float16 v8h;
typedef __attribute__((ext_vector_type(8)))  float    v8f;

#define NIMG 800            // 25 positions * 32 episodes
#define NPOS 25             // 20 support + 5 query
#define BN_EPS_C 0.001f

// ---------------------------------------------------------------------------
// WMMA fragment index helpers (CDNA5 ISA 7.12.2, 16-bit A 16x32 / B 32x16)
// ---------------------------------------------------------------------------
__device__ __forceinline__ int klocalA(int lane, int j, int h) {
  int base = (lane < 16) ? 0 : 8;
  return (j < 4) ? (base + 2 * j + h) : (16 + base + 2 * (j - 4) + h);
}
__device__ __forceinline__ int klocalB(int lane, int j, int h) {
  int base = (lane < 16) ? 0 : 8;
  return (j < 4) ? (base + 2 * j + h) : (16 + base + 2 * (j - 4) + h);
}

// ---------------------------------------------------------------------------
// Weight pre-pack: B fragments in exact lane/VGPR layout, f16.
// Conv layout: [ocTile(4)][chunk(NCH)][lane(32)][e(16)]
// ---------------------------------------------------------------------------
__global__ void pack_conv_w(const float* __restrict__ W, _Float16* __restrict__ Bp, int IC) {
  int NCH = (IC == 3) ? 1 : 18;
  int total = 4 * NCH * 512;
  int idx = blockIdx.x * 256 + threadIdx.x;
  if (idx >= total) return;
  int e    = idx & 15;
  int lane = (idx >> 4) & 31;
  int c    = (idx >> 9) % NCH;
  int nt   = idx / (512 * NCH);
  int j = e >> 1, h = e & 1;
  int klb = klocalB(lane, j, h);
  int tap, icb; bool valid = true;
  if (IC == 3) { valid = (klb < 27); tap = valid ? klb / 3 : 0; icb = valid ? klb % 3 : 0; }
  else         { tap = c >> 1; icb = ((c & 1) << 5) + klb; }
  float wv = valid ? W[((size_t)tap * IC + icb) * 64 + nt * 16 + (lane & 15)] : 0.f;
  Bp[idx] = (_Float16)wv;
}

// Proj layout: [nTile(8)][ch(50)][lane(32)][e(16)]
__global__ void pack_proj_w(const float* __restrict__ Wx, _Float16* __restrict__ Bp) {
  int idx = blockIdx.x * 256 + threadIdx.x;
  if (idx >= 8 * 50 * 512) return;
  int e    = idx & 15;
  int lane = (idx >> 4) & 31;
  int ch   = (idx >> 9) % 50;
  int nt   = idx / (512 * 50);
  int j = e >> 1, h = e & 1;
  int kl = klocalB(lane, j, h);
  Bp[idx] = (_Float16)Wx[(size_t)(ch * 32 + kl) * 128 + nt * 16 + (lane & 15)];
}

// ---------------------------------------------------------------------------
// Implicit-im2col conv tile: 16 pixels x 16 out-channels, K = 9*IC.
// Bp_wave: packed weights for this oc-tile ([chunk][lane][16] f16).
// IC==64: per chunk the tap is fixed -> A is two contiguous v8h loads/lane.
// ---------------------------------------------------------------------------
template <int HW, int IC>
__device__ v8f conv_tile(const _Float16* __restrict__ in_img,
                         const _Float16* __restrict__ Bp_wave,
                         int p0, int lane) {
  v8f acc = {0.f, 0.f, 0.f, 0.f, 0.f, 0.f, 0.f, 0.f};
  const int nPix = HW * HW;
  const int NCH  = (IC == 3) ? 1 : 18;
  int m = lane & 15;
  int p = p0 + m;
  if (p >= nPix) p = nPix - 1;           // clamp loads; stores masked by caller
  int py = p / HW, px = p % HW;
  int base = (lane < 16) ? 0 : 8;

  for (int c = 0; c < NCH; ++c) {
    v16h a;
    if (IC == 3) {
#pragma unroll
      for (int e = 0; e < 16; ++e) {
        int j = e >> 1, h = e & 1;
        int kl = klocalA(lane, j, h);
        bool valid = (kl < 27);
        int tap = valid ? kl / 3 : 0;
        int ic  = valid ? kl % 3 : 0;
        int iy = py + tap / 3 - 1;
        int ix = px + tap % 3 - 1;
        _Float16 av = (_Float16)0.f;
        if (valid && iy >= 0 && iy < HW && ix >= 0 && ix < HW)
          av = in_img[((size_t)iy * HW + ix) * IC + ic];
        a[e] = av;
      }
    } else {
      int tap = c >> 1;
      int icb = (c & 1) << 5;
      int iy = py + tap / 3 - 1;
      int ix = px + tap % 3 - 1;
      v8h alo = {}, ahi = {};
      if (iy >= 0 && iy < HW && ix >= 0 && ix < HW) {
        const _Float16* src = in_img + ((size_t)iy * HW + ix) * 64 + icb + base;
        alo = *(const v8h*)(src);
        ahi = *(const v8h*)(src + 16);
      }
      // Prefetch next tap's pixel row (gfx1250 global_prefetch path)
      if ((c & 1) && c + 1 < NCH) {
        int tapn = (c + 1) >> 1;
        int iyn = py + tapn / 3 - 1;
        int ixn = px + tapn % 3 - 1;
        if (iyn >= 0 && iyn < HW && ixn >= 0 && ixn < HW)
          __builtin_prefetch(in_img + ((size_t)iyn * HW + ixn) * 64, 0, 3);
      }
#pragma unroll
      for (int e = 0; e < 8; ++e) { a[e] = alo[e]; a[e + 8] = ahi[e]; }
    }
    v16h b = *(const v16h*)(Bp_wave + ((size_t)c * 32 + lane) * 16);
    acc = __builtin_amdgcn_wmma_f32_16x16x32_f16(false, a, false, b,
                                                 (short)0, acc, false, false);
  }
  return acc;
}

// ---------------------------------------------------------------------------
// Pack (B,S|Q,84,84,3) f32 inputs into [pos][b][84*84*3] f16
// ---------------------------------------------------------------------------
__global__ void pack_input(const float* __restrict__ xs, const float* __restrict__ xq,
                           _Float16* __restrict__ act0) {
  const int per = 84 * 84 * 3;
  size_t idx = (size_t)blockIdx.x * 256 + threadIdx.x;
  if (idx >= (size_t)NIMG * per) return;
  int img = (int)(idx / per);
  int off = (int)(idx % per);
  int pos = img / 32, b = img % 32;
  float v;
  if (pos < 20) v = xs[((size_t)b * 20 + pos) * per + off];
  else          v = xq[((size_t)b * 5 + (pos - 20)) * per + off];
  act0[idx] = (_Float16)v;
}

__global__ void zero_f32(float* p, int n) {
  int i = blockIdx.x * 256 + threadIdx.x;
  if (i < n) p[i] = 0.f;
}

// ---------------------------------------------------------------------------
// Pass A: conv + bias, accumulate per-(position,channel) sum / sum^2 for BN
// ---------------------------------------------------------------------------
template <int HW, int IC>
__global__ void conv_stats(const _Float16* __restrict__ act_in,
                           const _Float16* __restrict__ Bp, const float* __restrict__ bias,
                           float* __restrict__ sums, float* __restrict__ sqsums) {
  __shared__ float ssum[64], ssq[64];
  const int NCH = (IC == 3) ? 1 : 18;
  int img  = blockIdx.x;
  int wave = threadIdx.x >> 5, lane = threadIdx.x & 31;
  int n0 = (wave & 3) * 16;
  int p0 = blockIdx.y * 32 + (wave >> 2) * 16;
  const int nPix = HW * HW;
  if (threadIdx.x < 64) { ssum[threadIdx.x] = 0.f; ssq[threadIdx.x] = 0.f; }
  __syncthreads();

  const _Float16* in_img  = act_in + (size_t)img * nPix * IC;
  const _Float16* Bp_wave = Bp + (size_t)(wave & 3) * NCH * 512;
  if (p0 < nPix) {
    v8f acc = conv_tile<HW, IC>(in_img, Bp_wave, p0, lane);
    int n = lane & 15;
    float bv = bias[n0 + n];
    float s = 0.f, q = 0.f;
#pragma unroll
    for (int v = 0; v < 8; ++v) {
      int mrow = v + ((lane < 16) ? 0 : 8);
      if (p0 + mrow < nPix) { float x = acc[v] + bv; s += x; q += x * x; }
    }
    atomicAdd(&ssum[n0 + n], s);
    atomicAdd(&ssq[n0 + n], q);
  }
  __syncthreads();
  if (threadIdx.x < 64) {
    int g = img >> 5;   // position group
    atomicAdd(&sums[g * 64 + threadIdx.x],   ssum[threadIdx.x]);
    atomicAdd(&sqsums[g * 64 + threadIdx.x], ssq[threadIdx.x]);
  }
}

__global__ void finalize_stats(const float* __restrict__ sums, const float* __restrict__ sqsums,
                               const float* __restrict__ g, const float* __restrict__ be,
                               float count, float* __restrict__ scale, float* __restrict__ shift) {
  int i = blockIdx.x * 256 + threadIdx.x;
  if (i >= NPOS * 64) return;
  int ch = i & 63;
  float mean = sums[i] / count;
  float var  = sqsums[i] / count - mean * mean;
  float sc = g[ch] * rsqrtf(var + BN_EPS_C);
  scale[i] = sc;
  shift[i] = be[ch] - mean * sc;
}

// ---------------------------------------------------------------------------
// Pass B: conv + bias + BN + ReLU into LDS tile, 2x2 maxpool -> f16 out
// Block covers conv rows {2*r0, 2*r0+1} x cols [c0, c0+32) x 64 channels.
// ---------------------------------------------------------------------------
template <int HW, int IC>
__global__ void conv_bn_pool(const _Float16* __restrict__ act_in,
                             const _Float16* __restrict__ Bp, const float* __restrict__ bias,
                             const float* __restrict__ scale, const float* __restrict__ shift,
                             _Float16* __restrict__ act_out) {
  const int HWP = HW / 2;
  const int NCH = (IC == 3) ? 1 : 18;
  __shared__ float tile[2][32][64];
  int img = blockIdx.x, grp = img >> 5;
  int r0  = blockIdx.y;
  int c0  = blockIdx.z * 32;
  int wave = threadIdx.x >> 5, lane = threadIdx.x & 31;
  int n0 = (wave & 3) * 16, n = lane & 15;
  const _Float16* in_img  = act_in + (size_t)img * HW * HW * IC;
  const _Float16* Bp_wave = Bp + (size_t)(wave & 3) * NCH * 512;
  float sc = scale[grp * 64 + n0 + n];
  float sh = shift[grp * 64 + n0 + n];
  float bv = bias[n0 + n];

  for (int tsub = (wave >> 2); tsub < 4; tsub += 2) {
    int row  = 2 * r0 + (tsub >> 1);
    int colb = c0 + 16 * (tsub & 1);
    if (colb < HW) {
      int p0 = row * HW + colb;
      v8f acc = conv_tile<HW, IC>(in_img, Bp_wave, p0, lane);
#pragma unroll
      for (int v = 0; v < 8; ++v) {
        int mrow = v + ((lane < 16) ? 0 : 8);
        int col = colb + mrow;
        if (col < HW) {
          float x = (acc[v] + bv) * sc + sh;
          tile[tsub >> 1][col - c0][n0 + n] = fmaxf(x, 0.f);
        }
      }
    }
  }
  __syncthreads();

  _Float16* out_img = act_out + (size_t)img * HWP * HWP * 64;
  for (int i = threadIdx.x; i < 1024; i += 256) {
    int cx = i >> 6, ch = i & 63;
    int pc = (c0 >> 1) + cx;
    if (pc < HWP) {
      float a0 = tile[0][2 * cx][ch],     a1 = tile[0][2 * cx + 1][ch];
      float a2 = tile[1][2 * cx][ch],     a3 = tile[1][2 * cx + 1][ch];
      out_img[((size_t)r0 * HWP + pc) * 64 + ch] =
          (_Float16)fmaxf(fmaxf(a0, a1), fmaxf(a2, a3));
    }
  }
}

// ---------------------------------------------------------------------------
// LSTM input projection: proj[800][128] = emb[800][1600] @ Wx[1600][128] + b
// A rows are contiguous f16 -> two v8h loads; B pre-packed f16 fragments.
// ---------------------------------------------------------------------------
__global__ void proj_gemm(const _Float16* __restrict__ emb, const _Float16* __restrict__ Bp,
                          const float* __restrict__ bvec, float* __restrict__ proj) {
  int wave = threadIdx.x >> 5, lane = threadIdx.x & 31;
  int n0 = wave * 16;
  int m0 = blockIdx.x * 16;
  int m = lane & 15;
  int base = (lane < 16) ? 0 : 8;
  v8f acc = {0.f, 0.f, 0.f, 0.f, 0.f, 0.f, 0.f, 0.f};
  const _Float16* arow    = emb + (size_t)(m0 + m) * 1600;
  const _Float16* Bp_wave = Bp + (size_t)wave * 50 * 512;
  for (int ch = 0; ch < 50; ++ch) {
    v8h alo = *(const v8h*)(arow + ch * 32 + base);
    v8h ahi = *(const v8h*)(arow + ch * 32 + 16 + base);
    v16h a;
#pragma unroll
    for (int e = 0; e < 8; ++e) { a[e] = alo[e]; a[e + 8] = ahi[e]; }
    v16h b = *(const v16h*)(Bp_wave + ((size_t)ch * 32 + lane) * 16);
    acc = __builtin_amdgcn_wmma_f32_16x16x32_f16(false, a, false, b,
                                                 (short)0, acc, false, false);
  }
  float bb = bvec[n0 + m];
#pragma unroll
  for (int v = 0; v < 8; ++v) {
    int mr = v + ((lane < 16) ? 0 : 8);
    proj[(size_t)(m0 + mr) * 128 + n0 + m] = acc[v] + bb;
  }
}

// ---------------------------------------------------------------------------
// LSTM recurrence: grid(Q=5, dir=2), block(32 hidden, 21 rows).
// Time axis = episode axis (32 steps), per the reference.
// ---------------------------------------------------------------------------
__global__ void lstm_kernel(const float* __restrict__ projF, const float* __restrict__ projB,
                            const float* __restrict__ fr, const float* __restrict__ br,
                            float* __restrict__ outbuf) {
  __shared__ float hbuf[21][32];
  __shared__ float Whs[32][128];
  int q = blockIdx.x, dir = blockIdx.y;
  const float* proj = dir ? projB : projF;
  const float* Wh   = dir ? br : fr;
  int j = threadIdx.x, r = threadIdx.y;
  int tid = r * 32 + j;
  for (int i = tid; i < 32 * 128; i += 672) Whs[i >> 7][i & 127] = Wh[i];
  hbuf[r][j] = 0.f;
  float c = 0.f;
  int pos = (r < 20) ? r : (20 + q);
  __syncthreads();

  for (int step = 0; step < 32; ++step) {
    int tt = dir ? (31 - step) : step;
    const float* zrow = proj + (size_t)(pos * 32 + tt) * 128;
    float z[4];
#pragma unroll
    for (int g2 = 0; g2 < 4; ++g2) {
      float a = zrow[g2 * 32 + j];
      for (int k = 0; k < 32; ++k) a += hbuf[r][k] * Whs[k][g2 * 32 + j];
      z[g2] = a;
    }
    __syncthreads();
    float ig = 1.f / (1.f + __expf(-z[0]));
    float fg = 1.f / (1.f + __expf(-z[1]));
    float gg = tanhf(z[2]);
    float og = 1.f / (1.f + __expf(-z[3]));
    c = fg * c + ig * gg;
    float h = og * tanhf(c);
    hbuf[r][j] = h;
    outbuf[(((size_t)(q * 2 + dir) * 21 + r) * 32 + tt) * 32 + j] = h;
    __syncthreads();
  }
}

// ---------------------------------------------------------------------------
// Matching head: cosine attention over supports, softmax, CE + accuracy
// ---------------------------------------------------------------------------
__global__ void match_kernel(const float* __restrict__ outbuf,
                             const int* __restrict__ y_support, const int* __restrict__ y_query,
                             float* __restrict__ d_out) {
  __shared__ float ce_s[160], eq_s[160];
  int tid = threadIdx.x;
  if (tid < 160) {
    int q = tid / 32, b = tid % 32;
    float score[20];
    for (int s = 0; s < 20; ++s) {
      float dot = 0.f, ss = 0.f;
      for (int j = 0; j < 32; ++j) {
        size_t qf_i = (((size_t)(q * 2 + 0) * 21 + 20) * 32 + b) * 32 + j;
        size_t qb_i = (((size_t)(q * 2 + 1) * 21 + 20) * 32 + b) * 32 + j;
        size_t sf_i = (((size_t)(q * 2 + 0) * 21 + s)  * 32 + b) * 32 + j;
        size_t sb_i = (((size_t)(q * 2 + 1) * 21 + s)  * 32 + b) * 32 + j;
        float qf = outbuf[qf_i], qb = outbuf[qb_i];
        float sf = outbuf[sf_i], sb = outbuf[sb_i];
        dot += qf * sf + qb * sb;
        ss  += sf * sf + sb * sb;
      }
      score[s] = dot * rsqrtf(fmaxf(ss, 1e-10f));
    }
    float mx = score[0];
    for (int s = 1; s < 20; ++s) mx = fmaxf(mx, score[s]);
    float sum = 0.f;
    for (int s = 0; s < 20; ++s) { score[s] = __expf(score[s] - mx); sum += score[s]; }
    float inv = 1.f / sum;
    float preds[20];
    for (int w = 0; w < 20; ++w) preds[w] = 0.f;
    for (int s = 0; s < 20; ++s) preds[y_support[b * 20 + s]] += score[s] * inv;
    int yqv = y_query[b * 5 + q];
    float p = fminf(fmaxf(preds[yqv], 1e-7f), 1.f - 1e-7f);
    ce_s[tid] = -__logf(p);
    int best = 0; float bv = preds[0];
    for (int w = 1; w < 20; ++w) if (preds[w] > bv) { bv = preds[w]; best = w; }
    eq_s[tid] = (best == yqv) ? 1.f : 0.f;
  }
  __syncthreads();
  if (tid < 32) {
    float s = 0.f;
    for (int q = 0; q < 5; ++q) s += ce_s[q * 32 + tid];
    d_out[tid] = s * 0.2f;
  }
  if (tid == 0) {
    float a = 0.f;
    for (int i = 0; i < 160; ++i) a += eq_s[i];
    d_out[32] = a / 160.f;
  }
}

// ---------------------------------------------------------------------------
// Host orchestration
// ---------------------------------------------------------------------------
extern "C" void kernel_launch(void* const* d_in, const int* in_sizes, int n_in,
                              void* d_out, int out_size, void* d_ws, size_t ws_size,
                              hipStream_t stream) {
  (void)in_sizes; (void)n_in; (void)out_size; (void)ws_size;
  const float* xs  = (const float*)d_in[0];
  const int*   ysup= (const int*)  d_in[1];
  const float* xq  = (const float*)d_in[2];
  const int*   yqr = (const int*)  d_in[3];
  const float* k[4]  = {(const float*)d_in[4],  (const float*)d_in[8],
                        (const float*)d_in[12], (const float*)d_in[16]};
  const float* bi[4] = {(const float*)d_in[5],  (const float*)d_in[9],
                        (const float*)d_in[13], (const float*)d_in[17]};
  const float* gm[4] = {(const float*)d_in[6],  (const float*)d_in[10],
                        (const float*)d_in[14], (const float*)d_in[18]};
  const float* bt[4] = {(const float*)d_in[7],  (const float*)d_in[11],
                        (const float*)d_in[15], (const float*)d_in[19]};
  const float* fk = (const float*)d_in[20];
  const float* fr = (const float*)d_in[21];
  const float* fb = (const float*)d_in[22];
  const float* bk = (const float*)d_in[23];
  const float* br = (const float*)d_in[24];
  const float* bb = (const float*)d_in[25];
  float* out = (float*)d_out;

  // Workspace layout
  char* ws = (char*)d_ws;
  size_t off = 0;
  auto take = [&](size_t bytes) { size_t r = off; off = (off + bytes + 255) & ~(size_t)255; return r; };
  _Float16* act0 = (_Float16*)(ws + take((size_t)NIMG * 84 * 84 * 3 * 2));
  _Float16* act1 = (_Float16*)(ws + take((size_t)NIMG * 42 * 42 * 64 * 2));
  _Float16* act2 = (_Float16*)(ws + take((size_t)NIMG * 21 * 21 * 64 * 2));
  _Float16* act3 = (_Float16*)(ws + take((size_t)NIMG * 10 * 10 * 64 * 2));
  _Float16* act4 = (_Float16*)(ws + take((size_t)NIMG * 5  * 5  * 64 * 2));
  float* sums   = (float*)(ws + take(NPOS * 64 * 2 * sizeof(float)));   // sums + sqsums contiguous
  float* sqsums = sums + NPOS * 64;
  float* scale  = (float*)(ws + take(NPOS * 64 * sizeof(float)));
  float* shift  = (float*)(ws + take(NPOS * 64 * sizeof(float)));
  float* projF  = (float*)(ws + take((size_t)NIMG * 128 * sizeof(float)));
  float* projB  = (float*)(ws + take((size_t)NIMG * 128 * sizeof(float)));
  float* lstm_o = (float*)(ws + take((size_t)5 * 2 * 21 * 32 * 32 * sizeof(float)));
  _Float16* bpConv  = (_Float16*)(ws + take((size_t)4 * 18 * 512 * 2));   // max conv pack
  _Float16* bpProjF = (_Float16*)(ws + take((size_t)8 * 50 * 512 * 2));
  _Float16* bpProjB = (_Float16*)(ws + take((size_t)8 * 50 * 512 * 2));

  // Stage 0: pack inputs -> f16 [pos][b][84][84][3]
  {
    size_t total = (size_t)NIMG * 84 * 84 * 3;
    pack_input<<<dim3((unsigned)((total + 255) / 256)), 256, 0, stream>>>(xs, xq, act0);
  }

  // Conv blocks (pack weights -> pass A stats -> finalize -> pass B fused)
  const int nzero = NPOS * 64 * 2;
#define RUN_LAYER(HWv, ICv, INBUF, OUTBUF, LIDX, NTILES, HWPv, NSTRIP)                     \
  do {                                                                                     \
    int packN = 4 * ((ICv == 3) ? 1 : 18) * 512;                                           \
    pack_conv_w<<<dim3((packN + 255) / 256), 256, 0, stream>>>(k[LIDX], bpConv, ICv);      \
    zero_f32<<<dim3((nzero + 255) / 256), 256, 0, stream>>>(sums, nzero);                  \
    conv_stats<HWv, ICv><<<dim3(NIMG, NTILES), 256, 0, stream>>>(                          \
        INBUF, bpConv, bi[LIDX], sums, sqsums);                                            \
    finalize_stats<<<dim3((NPOS * 64 + 255) / 256), 256, 0, stream>>>(                     \
        sums, sqsums, gm[LIDX], bt[LIDX], (float)(32.0 * HWv * HWv), scale, shift);        \
    conv_bn_pool<HWv, ICv><<<dim3(NIMG, HWPv, NSTRIP), 256, 0, stream>>>(                  \
        INBUF, bpConv, bi[LIDX], scale, shift, OUTBUF);                                    \
  } while (0)

  RUN_LAYER(84, 3,  act0, act1, 0, 221, 42, 3);
  RUN_LAYER(42, 64, act1, act2, 1, 56,  21, 2);
  RUN_LAYER(21, 64, act2, act3, 2, 14,  10, 1);
  RUN_LAYER(10, 64, act3, act4, 3, 4,   5,  1);
#undef RUN_LAYER

  // LSTM input projections (both directions), packed-B GEMMs
  pack_proj_w<<<dim3((8 * 50 * 512 + 255) / 256), 256, 0, stream>>>(fk, bpProjF);
  pack_proj_w<<<dim3((8 * 50 * 512 + 255) / 256), 256, 0, stream>>>(bk, bpProjB);
  proj_gemm<<<dim3(50), 256, 0, stream>>>(act4, bpProjF, fb, projF);
  proj_gemm<<<dim3(50), 256, 0, stream>>>(act4, bpProjB, bb, projB);

  // Bidirectional LSTM recurrence
  lstm_kernel<<<dim3(5, 2), dim3(32, 21, 1), 0, stream>>>(projF, projB, fr, br, lstm_o);

  // Matching head -> ce_out[32], acc
  match_kernel<<<dim3(1), 256, 0, stream>>>(lstm_o, ysup, yqr, out);
}
